// AttentionComponent_15960098472670
// MI455X (gfx1250) — matrix operand order repaired and checked by reading, not compiled, not run on hardware
//
#include <hip/hip_runtime.h>

#define BATCH  4
#define SEQ    2048
#define DMODEL 1024
#define CDIM   128

typedef __attribute__((ext_vector_type(16))) __bf16 v16bf;
typedef __attribute__((ext_vector_type(8)))  __bf16 v8bf;
typedef __attribute__((ext_vector_type(8)))  float  v8f;

union BF16x16 { v16bf v; v8bf h2[2]; unsigned u[8]; };

// fp32 -> bf16 RNE
static __device__ __forceinline__ unsigned rne(float f) {
    unsigned u = __builtin_bit_cast(unsigned, f);
    return u + 0x7FFFu + ((u >> 16) & 1u);
}
static __device__ __forceinline__ __bf16 cvt_bf16(float f) {
    unsigned short hs = (unsigned short)(rne(f) >> 16);
    return __builtin_bit_cast(__bf16, hs);
}

// A-fragment element i -> K = 16*h + i: one contiguous run of 16
static __device__ __forceinline__ v16bf load_afrag_bf16(const __bf16* rowp, int h) {
    BF16x16 r;
    r.h2[0] = *(const v8bf*)(rowp + 16 * h);
    r.h2[1] = *(const v8bf*)(rowp + 16 * h + 8);
    return r.v;
}
// B-fragment element i -> K = 16*h + i: one contiguous run of 16 (pass p already +16h)
static __device__ __forceinline__ v16bf load_bfrag_bf16(const __bf16* p) {
    BF16x16 r;
    r.h2[0] = *(const v8bf*)(p);
    r.h2[1] = *(const v8bf*)(p + 8);
    return r.v;
}

static __device__ __forceinline__ v8f wmma_bf16(v16bf a, v16bf b, v8f c) {
    return __builtin_amdgcn_wmma_f32_16x16x32_bf16(
        false, a, false, b, (short)0, c, false, false);
}

// ---------------------------------------------------------------------------
// Prep A: dstT[c*R + r] = bf16(src[r*Cc + c]); optional row-major bf16 copy.
// ---------------------------------------------------------------------------
__global__ void tconv_kernel(const float* __restrict__ src,
                             __bf16* __restrict__ dstT,
                             __bf16* __restrict__ dstR, int R, int Cc) {
    int i = blockIdx.x * blockDim.x + threadIdx.x;
    if (i >= R * Cc) return;
    int r = i / Cc, c = i - r * Cc;
    __bf16 v = cvt_bf16(src[i]);
    dstT[(size_t)c * R + r] = v;
    if (dstR) dstR[i] = v;
}
// Prep B: plain fp32 -> bf16 convert
__global__ void conv_kernel(const float* __restrict__ src,
                            __bf16* __restrict__ dst, int n) {
    int i = blockIdx.x * blockDim.x + threadIdx.x;
    if (i < n) dst[i] = cvt_bf16(src[i]);
}

// ---------------------------------------------------------------------------
// Kernel 1: qm = (x @ A) * mask (bf16),  kk = x @ Bmat^T (bf16)
// All operands pre-converted bf16, K contiguous -> pure b128 loads + WMMA.
// ---------------------------------------------------------------------------
__global__ void proj_kernel(const __bf16* __restrict__ xb,    // [B*S, D]
                            const __bf16* __restrict__ Atb,   // [C, D]
                            const __bf16* __restrict__ Bmb,   // [C, D]
                            const float* __restrict__ mask,
                            __bf16* __restrict__ qm,
                            __bf16* __restrict__ kk) {
    const int wave  = blockIdx.x * (blockDim.x >> 5) + (threadIdx.x >> 5);
    const int lane  = threadIdx.x & 31;
    const int h     = lane >> 4;
    const int l15   = lane & 15;
    const int ntile = wave & 7;      // CDIM/16
    const int mtile = wave >> 3;     // (BATCH*SEQ)/16 = 512
    const int row0  = mtile * 16;
    const int col0  = ntile * 16;

    const __bf16* xrow = xb  + (size_t)(row0 + l15) * DMODEL;
    const __bf16* arow = Atb + (size_t)(col0 + l15) * DMODEL + 16 * h;
    const __bf16* brow = Bmb + (size_t)(col0 + l15) * DMODEL + 16 * h;

    v8f accq = {};
    v8f acck = {};
    for (int k0 = 0; k0 < DMODEL; k0 += 32) {
        __builtin_prefetch(xrow + k0 + 128, 0, 1);
        v16bf ax = load_afrag_bf16(xrow + k0, h);
        v16bf bq = load_bfrag_bf16(arow + k0);
        v16bf bk = load_bfrag_bf16(brow + k0);
        accq = wmma_bf16(ax, bq, accq);
        acck = wmma_bf16(ax, bk, acck);
    }
    #pragma unroll
    for (int v = 0; v < 8; ++v) {
        int    r   = row0 + v + (h << 3);
        size_t idx = (size_t)r * CDIM + (col0 + l15);
        qm[idx] = cvt_bf16(accq[v] * mask[idx]);
        kk[idx] = cvt_bf16(acck[v]);
    }
}

// ---------------------------------------------------------------------------
// Kernel 2: flash attention.  One wave: 16 query rows x 128 D-columns.
// ---------------------------------------------------------------------------
__global__ void attn_kernel(const __bf16* __restrict__ xTb,   // [B, D, S]
                            const __bf16* __restrict__ qm,
                            const __bf16* __restrict__ kk,
                            __bf16* __restrict__ zb) {
    __shared__ __align__(16) __bf16 Pl[4 * 16 * 32];
    const int wid  = threadIdx.x >> 5;
    const int wave = blockIdx.x * (blockDim.x >> 5) + wid;
    const int lane = threadIdx.x & 31;
    const int h    = lane >> 4;
    const int l15  = lane & 15;

    const int dseg  = wave & 7;
    const int qtile = (wave >> 3) & 127;
    const int b     = wave >> 10;
    const int q0    = qtile * 16;
    const int dbase = dseg * 128;
    __bf16* P = Pl + wid * (16 * 32);

    // Preload qm A-fragments (K = C = 128 -> 4 chunks of 32)
    v16bf qa[4];
    #pragma unroll
    for (int ch = 0; ch < 4; ++ch)
        qa[ch] = load_afrag_bf16(
            qm + ((size_t)b * SEQ + q0 + l15) * CDIM + ch * 32, h);

    v8f acc[8];
    #pragma unroll
    for (int f = 0; f < 8; ++f) acc[f] = (v8f){};
    float mstat[8], lstat[8];
    #pragma unroll
    for (int v = 0; v < 8; ++v) { mstat[v] = -3.0e38f; lstat[v] = 0.0f; }

    const int nblk = (q0 + 16 + 31) >> 5;
    for (int blk = 0; blk < nblk; ++blk) {
        float p[2][8];
        #pragma unroll
        for (int j = 0; j < 2; ++j) {
            const int kb = blk * 32 + j * 16;
            const __bf16* krow = kk + ((size_t)b * SEQ + kb + l15) * CDIM + 16 * h;
            v8f s = {};
            #pragma unroll
            for (int ch = 0; ch < 4; ++ch)
                s = wmma_bf16(qa[ch], load_bfrag_bf16(krow + ch * 32), s);
            const int kcol = kb + l15;
            #pragma unroll
            for (int v = 0; v < 8; ++v) {
                int   qrow = q0 + v + (h << 3);
                float val  = s[v] * (1.0f / (float)DMODEL);
                if (kcol > qrow) val = -3.0e38f;     // causal mask
                p[j][v] = val;
            }
        }
        // Online softmax: row r = v + 8h lives across 16 lanes of one half.
        #pragma unroll
        for (int v = 0; v < 8; ++v) {
            float mx = fmaxf(p[0][v], p[1][v]);
            #pragma unroll
            for (int off = 1; off < 16; off <<= 1)
                mx = fmaxf(mx, __shfl_xor(mx, off, 32));
            float mnew  = fmaxf(mstat[v], mx);
            float scale = __expf(mstat[v] - mnew);
            mstat[v] = mnew;
            lstat[v] *= scale;
            #pragma unroll
            for (int f = 0; f < 8; ++f) acc[f][v] *= scale;
            p[0][v] = __expf(p[0][v] - mnew);
            p[1][v] = __expf(p[1][v] - mnew);
            float rs = p[0][v] + p[1][v];
            #pragma unroll
            for (int off = 1; off < 16; off <<= 1)
                rs += __shfl_xor(rs, off, 32);
            lstat[v] += rs;
        }
        // Route P through LDS: C/D layout -> A-fragment layout.
        #pragma unroll
        for (int j = 0; j < 2; ++j)
            #pragma unroll
            for (int v = 0; v < 8; ++v)
                P[(v + (h << 3)) * 32 + j * 16 + l15] = cvt_bf16(p[j][v]);
        asm volatile("s_wait_dscnt 0" ::: "memory");
        v16bf pa = load_afrag_bf16(P + l15 * 32, h);
        asm volatile("s_wait_dscnt 0" ::: "memory");
        // z += P @ x : B[K=key][N=col] = xTb[b, col, key], key contiguous
        #pragma unroll
        for (int f = 0; f < 8; ++f) {
            const __bf16* xc = xTb +
                ((size_t)b * DMODEL + dbase + f * 16 + l15) * SEQ + blk * 32 + 16 * h;
            acc[f] = wmma_bf16(pa, load_bfrag_bf16(xc), acc[f]);
        }
    }
    #pragma unroll
    for (int v = 0; v < 8; ++v) {
        float inv  = 1.0f / lstat[v];
        int   qrow = q0 + v + (h << 3);
        __bf16* zr = zb + ((size_t)b * SEQ + qrow) * DMODEL + dbase;
        #pragma unroll
        for (int f = 0; f < 8; ++f) zr[f * 16 + l15] = cvt_bf16(acc[f][v] * inv);
    }
}

// ---------------------------------------------------------------------------
// Kernel 3: out = z @ ov (fp32 out), ov pre-transposed to [E, D]
// ---------------------------------------------------------------------------
__global__ void out_kernel(const __bf16* __restrict__ zb,
                           const __bf16* __restrict__ ovTb,   // [E, D]
                           float* __restrict__ out) {
    const int wave  = blockIdx.x * (blockDim.x >> 5) + (threadIdx.x >> 5);
    const int lane  = threadIdx.x & 31;
    const int h     = lane >> 4;
    const int l15   = lane & 15;
    const int ntile = wave & 63;
    const int mtile = wave >> 6;
    const int row0  = mtile * 16;
    const int col0  = ntile * 16;

    const __bf16* zrow = zb   + (size_t)(row0 + l15) * DMODEL;
    const __bf16* orow = ovTb + (size_t)(col0 + l15) * DMODEL + 16 * h;

    v8f acc = {};
    for (int k0 = 0; k0 < DMODEL; k0 += 32) {
        __builtin_prefetch(zrow + k0 + 128, 0, 1);
        v16bf az = load_afrag_bf16(zrow + k0, h);
        v16bf bo = load_bfrag_bf16(orow + k0);
        acc = wmma_bf16(az, bo, acc);
    }
    #pragma unroll
    for (int v = 0; v < 8; ++v)
        out[(size_t)(row0 + v + (h << 3)) * DMODEL + col0 + l15] = acc[v];
}

// ---------------------------------------------------------------------------
extern "C" void kernel_launch(void* const* d_in, const int* in_sizes, int n_in,
                              void* d_out, int out_size, void* d_ws, size_t ws_size,
                              hipStream_t stream) {
    const float* x    = (const float*)d_in[0];
    const float* A    = (const float*)d_in[1];
    const float* Bm   = (const float*)d_in[2];
    const float* ov   = (const float*)d_in[3];
    const float* mask = (const float*)d_in[4];

    char* ws = (char*)d_ws;
    const size_t MB = 1024 * 1024;
    __bf16* qm   = (__bf16*)(ws);                       // 2 MB
    __bf16* kk   = (__bf16*)(ws + 2 * MB);              // 2 MB
    __bf16* zb   = (__bf16*)(ws + 4 * MB);              // 16 MB
    __bf16* xTb  = (__bf16*)(ws + 20 * MB);             // 16 MB  [B, D, S]
    __bf16* xb   = (__bf16*)(ws + 36 * MB);             // 16 MB  [B*S, D]
    __bf16* Atb  = (__bf16*)(ws + 52 * MB);             // 256 KB [C, D]
    __bf16* Bmb  = (__bf16*)(ws + 52 * MB + 512 * 1024);// 256 KB [C, D]
    __bf16* ovTb = (__bf16*)(ws + 53 * MB);             // 2 MB   [E, D]
    float*  out  = (float*)d_out;

    // Prep: all bf16 operands, K-contiguous where used as B operands.
    tconv_kernel<<<512,  256, 0, stream>>>(A,  Atb,  (__bf16*)nullptr, DMODEL, CDIM);
    tconv_kernel<<<4096, 256, 0, stream>>>(ov, ovTb, (__bf16*)nullptr, DMODEL, DMODEL);
    conv_kernel <<<512,  256, 0, stream>>>(Bm, Bmb, CDIM * DMODEL);
    for (int b = 0; b < BATCH; ++b)
        tconv_kernel<<<8192, 256, 0, stream>>>(
            x + (size_t)b * SEQ * DMODEL, xTb + (size_t)b * DMODEL * SEQ,
            xb + (size_t)b * SEQ * DMODEL, SEQ, DMODEL);

    proj_kernel<<<1024, 128, 0, stream>>>(xb, Atb, Bmb, mask, qm, kk);
    attn_kernel<<<1024, 128, 0, stream>>>(xTb, qm, kk, zb);
    out_kernel <<<8192, 128, 0, stream>>>(zb, ovTb, out);
}
